// ArcNet_61589831025164
// MI455X (gfx1250) — compile-verified, matrix operand order checked by reading, not run on hardware
//
#include <hip/hip_runtime.h>
#include <hip/hip_bf16.h>
#include <math.h>

// ---------------- problem constants (from reference) ----------------
#define B_   1024
#define D_   512
#define C_   100000
#define S_   20.0f
#define MRG_ 0.1f
#define EPS_ 1e-7f

// ---------------- WMMA vector types ----------------
typedef __attribute__((ext_vector_type(16))) __bf16 v16bf;
typedef __attribute__((ext_vector_type(8)))  __bf16 v8bf;
typedef __attribute__((ext_vector_type(8)))  float  v8f;

// ---------------- async global->LDS copy (CDNA5), guarded ----------------
#if defined(__HIP_DEVICE_COMPILE__) && defined(__gfx1250__) && \
    __has_builtin(__builtin_amdgcn_global_load_async_to_lds_b128)
#define USE_ASYNC_COPY 1
#else
#define USE_ASYNC_COPY 0
#endif

typedef int v4i_ __attribute__((vector_size(16)));
#if USE_ASYNC_COPY
typedef __attribute__((address_space(1))) v4i_* as1_v4i;
typedef __attribute__((address_space(3))) v4i_* as3_v4i;
#endif

__device__ __forceinline__ void copy16(void* lds, const void* g) {
#if USE_ASYNC_COPY
  void* gnc = const_cast<void*>(g);
  __builtin_amdgcn_global_load_async_to_lds_b128((as1_v4i)gnc, (as3_v4i)lds, 0, 0);
#else
  *reinterpret_cast<uint4*>(lds) = *reinterpret_cast<const uint4*>(g);
#endif
}

// wait until <= 8 async ops outstanding (previous tile drained, next tile in flight)
__device__ __forceinline__ void stage_wait_prev() {
#if USE_ASYNC_COPY
#if __has_builtin(__builtin_amdgcn_s_wait_asynccnt)
  __builtin_amdgcn_s_wait_asynccnt(8);
#else
  asm volatile("s_wait_asynccnt 8" ::: "memory");
#endif
#endif
}
// wait until all async ops complete
__device__ __forceinline__ void stage_wait_all() {
#if USE_ASYNC_COPY
#if __has_builtin(__builtin_amdgcn_s_wait_asynccnt)
  __builtin_amdgcn_s_wait_asynccnt(0);
#else
  asm volatile("s_wait_asynccnt 0" ::: "memory");
#endif
#endif
}

// ==================== kernel 1: inverse row norms of W ====================
__global__ __launch_bounds__(256) void wnorm_kernel(const float* __restrict__ W,
                                                    float* __restrict__ invn) {
  const int wave = threadIdx.x >> 5;
  const int lane = threadIdx.x & 31;
  const int row  = blockIdx.x * 8 + wave;
  if (row >= C_) return;
  const float* p = W + (size_t)row * D_;
  float ss = 0.f;
  #pragma unroll 4
  for (int k = lane; k < D_; k += 32) { float v = p[k]; ss += v * v; }
  #pragma unroll
  for (int off = 16; off > 0; off >>= 1) ss += __shfl_xor(ss, off, 32);
  if (lane == 0) invn[row] = 1.f / fmaxf(sqrtf(ss), 1e-12f);
}

// ==================== kernel 2: split x into bf16 hi/lo ====================
__global__ __launch_bounds__(256) void xsplit_kernel(const float* __restrict__ x,
                                                     __bf16* __restrict__ xhi,
                                                     __bf16* __restrict__ xlo) {
  const int i = blockIdx.x * 256 + threadIdx.x;   // grid sized exactly B_*D_/256
  float v = x[i];
  __bf16 h = (__bf16)v;
  xhi[i] = h;
  xlo[i] = (__bf16)(v - (float)h);
}

// ==================== kernel 3: WMMA GEMM  logits = (x . W^T)*invn + b ====================
// Block tile: 128(M) x 128(N), K-step 32, double-buffered async LDS staging.
// 8 waves: wave_m(2) x wave_n(4); each wave owns 4x2 subtiles of 16x16 (v8f acc).
#define BM  128
#define BN  128
#define BK  32
#define NK  (D_ / BK)          // 16 K-steps
#define LDA 40                 // bf16 elems / row in LDS (32 + 8 pad -> 80B rows)
#define LDW 36                 // f32  elems / row in LDS (32 + 4 pad -> 144B rows)
#define ASZ (BM * LDA)         // one A buffer (bf16 elems)
#define WSZ (BN * LDW)         // one W buffer (f32 elems)

__device__ __forceinline__ v16bf frag_from_lds_bf16(const __bf16* p) {
  // p points at [khalf]; fragment = p[0..7] ++ p[16..23]  (CDNA5 16-bit A/B layout)
  v8bf c0 = *reinterpret_cast<const v8bf*>(p);
  v8bf c1 = *reinterpret_cast<const v8bf*>(p + 16);
  return __builtin_shufflevector(c0, c1, 0,1,2,3,4,5,6,7,8,9,10,11,12,13,14,15);
}

// issue one tile's staging: 8 x 16B async copies per thread
__device__ __forceinline__ void stage_tile(int tid, int row0, int col0, int k0,
                                           const __bf16* __restrict__ xhi,
                                           const __bf16* __restrict__ xlo,
                                           const float*  __restrict__ W,
                                           __bf16* sAhi, __bf16* sAlo, float* sW) {
  // A hi/lo: 128 rows x 64B each = 512 16B chunks per buffer
  #pragma unroll
  for (int q = 0; q < 2; ++q) {
    int cid = tid + q * 256;
    int r = cid >> 2, ch = cid & 3;
    size_t goff = (size_t)(row0 + r) * D_ + k0 + ch * 8;
    copy16(&sAhi[r * LDA + ch * 8], &xhi[goff]);
    copy16(&sAlo[r * LDA + ch * 8], &xlo[goff]);
  }
  // W (f32): 128 rows x 128B = 1024 16B chunks
  #pragma unroll
  for (int q = 0; q < 4; ++q) {
    int cid = tid + q * 256;
    int r = cid >> 3, ch = cid & 7;
    int cls = col0 + r; if (cls >= C_) cls = C_ - 1;   // clamp; masked at store
    copy16(&sW[r * LDW + ch * 4], &W[(size_t)cls * D_ + k0 + ch * 4]);
  }
}

__global__ __launch_bounds__(256) void gemm_kernel(const __bf16* __restrict__ xhi,
                                                   const __bf16* __restrict__ xlo,
                                                   const float*  __restrict__ W,
                                                   const float*  __restrict__ invn,
                                                   const float*  __restrict__ bias,
                                                   float* __restrict__ out) {
  __shared__ __align__(16) __bf16 sAhi[2 * ASZ];
  __shared__ __align__(16) __bf16 sAlo[2 * ASZ];
  __shared__ __align__(16) float  sW  [2 * WSZ];

  const int tid   = threadIdx.x;
  const int lane  = tid & 31;
  const int wave  = tid >> 5;
  const int wm    = wave >> 2;        // 0..1
  const int wn    = wave & 3;         // 0..3
  const int lrow  = lane & 15;
  const int lhalf = lane >> 4;        // 0/1
  const int khalf = lhalf * 8;

  const int row0 = blockIdx.y * BM;
  const int col0 = blockIdx.x * BN;

  v8f acc[4][2];
  const v8f vzero = {0.f,0.f,0.f,0.f,0.f,0.f,0.f,0.f};
  #pragma unroll
  for (int mt = 0; mt < 4; ++mt)
    #pragma unroll
    for (int nt = 0; nt < 2; ++nt) acc[mt][nt] = vzero;

  // prologue: prefetch tile 0 into buffer 0
  stage_tile(tid, row0, col0, 0, xhi, xlo, W, sAhi, sAlo, sW);

  for (int ks = 0; ks < NK; ++ks) {
    const int cur = ks & 1;
    if (ks + 1 < NK) {
      // overlap: launch tile ks+1 into the other buffer, then drain only tile ks
      const int nxt = cur ^ 1;
      stage_tile(tid, row0, col0, (ks + 1) * BK, xhi, xlo, W,
                 sAhi + nxt * ASZ, sAlo + nxt * ASZ, sW + nxt * WSZ);
      stage_wait_prev();     // s_wait_asynccnt 8: tile ks complete, ks+1 in flight
    } else {
      stage_wait_all();      // s_wait_asynccnt 0
    }
    __syncthreads();

    const __bf16* bAhi = sAhi + cur * ASZ;
    const __bf16* bAlo = sAlo + cur * ASZ;
    const float*  bW   = sW   + cur * WSZ;

    // B fragments: convert f32 -> bf16 hi/lo at load (w = hi + lo)
    v16bf bhi[2], blo[2];
    #pragma unroll
    for (int nt = 0; nt < 2; ++nt) {
      const float* p = &bW[(wn * 32 + nt * 16 + lrow) * LDW + khalf];
      #pragma unroll
      for (int i = 0; i < 8; ++i) {
        float v0 = p[i], v1 = p[i + 16];
        __bf16 h0 = (__bf16)v0, h1 = (__bf16)v1;
        bhi[nt][i]     = h0; blo[nt][i]     = (__bf16)(v0 - (float)h0);
        bhi[nt][i + 8] = h1; blo[nt][i + 8] = (__bf16)(v1 - (float)h1);
      }
    }
    #pragma unroll
    for (int mt = 0; mt < 4; ++mt) {
      const int ar = (wm * 64 + mt * 16 + lrow) * LDA + khalf;
      v16bf ahi = frag_from_lds_bf16(&bAhi[ar]);
      v16bf alo = frag_from_lds_bf16(&bAlo[ar]);
      #pragma unroll
      for (int nt = 0; nt < 2; ++nt) {
        // bf16x3: hi*hi + hi*lo + lo*hi  ~= fp32 dot product
        acc[mt][nt] = __builtin_amdgcn_wmma_f32_16x16x32_bf16(
            false, ahi, false, bhi[nt], (short)0, acc[mt][nt], false, false);
        acc[mt][nt] = __builtin_amdgcn_wmma_f32_16x16x32_bf16(
            false, ahi, false, blo[nt], (short)0, acc[mt][nt], false, false);
        acc[mt][nt] = __builtin_amdgcn_wmma_f32_16x16x32_bf16(
            false, alo, false, bhi[nt], (short)0, acc[mt][nt], false, false);
      }
    }
    __syncthreads();  // all waves done with buffer `cur` before it is restaged
  }

  // epilogue: wf = acc * invnorm[j] + b[j]
  #pragma unroll
  for (int nt = 0; nt < 2; ++nt) {
    const int j = col0 + wn * 32 + nt * 16 + lrow;
    if (j < C_) {
      const float inv = invn[j];
      const float bb  = bias[j];
      #pragma unroll
      for (int mt = 0; mt < 4; ++mt) {
        const int i0 = row0 + wm * 64 + mt * 16 + lhalf * 8;  // C/D layout: m = 8*lhalf + v
        #pragma unroll
        for (int v = 0; v < 8; ++v)
          out[(size_t)(i0 + v) * C_ + j] = acc[mt][nt][v] * inv + bb;
      }
    }
  }
}

// ==================== kernel 4: per-row online softmax stats ====================
// Tracks m = max(wf), s1 = sum exp(wf-m), sS = sum exp(S*(wf-m)); grabs target logit.
__global__ __launch_bounds__(256) void rowstats_kernel(const float* __restrict__ logits,
                                                       const int* __restrict__ labels,
                                                       float* __restrict__ rowmax,
                                                       float* __restrict__ rows1,
                                                       float* __restrict__ rowsS,
                                                       float* __restrict__ rowtgt) {
  const int row = blockIdx.x;
  const float* p = logits + (size_t)row * C_;
  float m = -3.4e38f, s1 = 0.f, sS = 0.f;
  for (int j = threadIdx.x; j < C_; j += 256) {
    float v = p[j];
    if (v > m) {
      s1 = s1 * __expf(m - v) + 1.f;
      sS = sS * __expf(S_ * (m - v)) + 1.f;
      m = v;
    } else {
      s1 += __expf(v - m);
      sS += __expf(S_ * (v - m));
    }
  }
  #pragma unroll
  for (int off = 16; off > 0; off >>= 1) {
    float m2  = __shfl_xor(m,  off, 32);
    float s12 = __shfl_xor(s1, off, 32);
    float sS2 = __shfl_xor(sS, off, 32);
    float mm  = fmaxf(m, m2);
    s1 = s1 * __expf(m - mm) + s12 * __expf(m2 - mm);
    sS = sS * __expf(S_ * (m - mm)) + sS2 * __expf(S_ * (m2 - mm));
    m = mm;
  }
  __shared__ float sm[8], ss1[8], ssS[8];
  const int wave = threadIdx.x >> 5, lane = threadIdx.x & 31;
  if (lane == 0) { sm[wave] = m; ss1[wave] = s1; ssS[wave] = sS; }
  __syncthreads();
  if (threadIdx.x == 0) {
    m = sm[0]; s1 = ss1[0]; sS = ssS[0];
    #pragma unroll
    for (int w = 1; w < 8; ++w) {
      float mm = fmaxf(m, sm[w]);
      s1 = s1 * __expf(m - mm) + ss1[w] * __expf(sm[w] - mm);
      sS = sS * __expf(S_ * (m - mm)) + ssS[w] * __expf(S_ * (sm[w] - mm));
      m = mm;
    }
    rowmax[row] = m; rows1[row] = s1; rowsS[row] = sS;
    rowtgt[row] = p[labels[row]];
  }
}

// ==================== kernel 5: ArcFace loss (stable log-domain) ====================
__global__ __launch_bounds__(256) void loss_kernel(const float* __restrict__ rowmax,
                                                   const float* __restrict__ rowsS,
                                                   const float* __restrict__ rowtgt,
                                                   float* __restrict__ loss_out) {
  float accv = 0.f;
  for (int row = threadIdx.x; row < B_; row += 256) {
    float m = rowmax[row], sS = rowsS[row], t = rowtgt[row];
    float tc  = fminf(fmaxf(t, -1.f + EPS_), 1.f - EPS_);
    float num = S_ * cosf(acosf(tc) + MRG_);
    float E   = sS - expf(S_ * (t - m));          // excl sum * exp(-S*m)
    E = fmaxf(E, 0.f);
    float mm = fmaxf(num, S_ * m);
    float L  = num - (mm + logf(expf(num - mm) + E * expf(S_ * m - mm)));
    accv += L;
  }
  #pragma unroll
  for (int off = 16; off > 0; off >>= 1) accv += __shfl_xor(accv, off, 32);
  __shared__ float sw[8];
  const int wave = threadIdx.x >> 5, lane = threadIdx.x & 31;
  if (lane == 0) sw[wave] = accv;
  __syncthreads();
  if (threadIdx.x == 0) {
    float tot = 0.f;
    #pragma unroll
    for (int w = 0; w < 8; ++w) tot += sw[w];
    loss_out[0] = -tot / (float)B_;
  }
}

// ==================== kernel 6: in-place softmax write ====================
__global__ __launch_bounds__(256) void softmax_kernel(float* __restrict__ out,
                                                      const float* __restrict__ rowmax,
                                                      const float* __restrict__ rows1) {
  const size_t i4 = (size_t)blockIdx.x * 256 + threadIdx.x;  // B_*C_/4 elems, C_%4==0
  const size_t e0 = i4 * 4;
  const int row = (int)(e0 / C_);
  const float m = rowmax[row];
  const float r = 1.f / rows1[row];
  float4 v = reinterpret_cast<float4*>(out)[i4];
  v.x = __expf(v.x - m) * r;
  v.y = __expf(v.y - m) * r;
  v.z = __expf(v.z - m) * r;
  v.w = __expf(v.w - m) * r;
  reinterpret_cast<float4*>(out)[i4] = v;
}

// ==================== launch ====================
extern "C" void kernel_launch(void* const* d_in, const int* in_sizes, int n_in,
                              void* d_out, int out_size, void* d_ws, size_t ws_size,
                              hipStream_t stream) {
  const float* x      = (const float*)d_in[0];
  const float* W      = (const float*)d_in[1];
  const float* b      = (const float*)d_in[2];
  const int*   labels = (const int*)d_in[3];
  float* out = (float*)d_out;

  // workspace carve (all offsets 128B aligned); total ~2.4 MB
  char* w = (char*)d_ws;
  float*  invn   = (float*)(w + 0);               // C_ floats (400000B -> pad 400384)
  __bf16* xhi    = (__bf16*)(w + 400384);         // B_*D_ bf16 = 1 MB
  __bf16* xlo    = (__bf16*)(w + 1448960);        // 1 MB
  float*  rowmax = (float*)(w + 2497536);         // B_ floats
  float*  rows1  = (float*)(w + 2501632);
  float*  rowsS  = (float*)(w + 2505728);
  float*  rowtgt = (float*)(w + 2509824);

  wnorm_kernel <<<(C_ + 7) / 8, 256, 0, stream>>>(W, invn);
  xsplit_kernel<<<(B_ * D_) / 256, 256, 0, stream>>>(x, xhi, xlo);
  gemm_kernel  <<<dim3((C_ + BN - 1) / BN, B_ / BM), 256, 0, stream>>>(
      xhi, xlo, W, invn, b, out);
  rowstats_kernel<<<B_, 256, 0, stream>>>(out, labels, rowmax, rows1, rowsS, rowtgt);
  loss_kernel  <<<1, 256, 0, stream>>>(rowmax, rowsS, rowtgt, out + (size_t)B_ * C_);
  softmax_kernel<<<(B_ * (C_ / 4)) / 256, 256, 0, stream>>>(out, rowmax, rows1);
}
